// Decoder_46926812677044
// MI455X (gfx1250) — matrix operand order repaired and checked by reading, not compiled, not run on hardware
//
#include <hip/hip_runtime.h>
#include <hip/hip_bf16.h>
#include <math.h>

// ---------------- problem constants ----------------
#define BATCH 64
#define SPOS  196          // 14*14 attention positions
#define ENCD  2048
#define LCAP  20
#define TSTEP 19           // L-1
#define ADIM  512
#define DDIM  512
#define EDIM  512
#define VOC   10000
#define XDIM  3072         // [emb(512) | ctx(2048) | h(512)]

typedef __attribute__((ext_vector_type(16))) _Float16 v16h;
typedef __attribute__((ext_vector_type(8)))  _Float16 h8;
typedef __attribute__((ext_vector_type(8)))  float    v8f;

__device__ __forceinline__ float sigf(float x) { return 1.0f / (1.0f + __expf(-x)); }

// ---------------------------------------------------------------------------
// Register-blocked WMMA GEMM:
//   C(MxN,f32) = A(MxK,f16 row-major, lda) * Bt(NxK,f16 "n-major", ldb) + bias[n]
// One wave computes a (16*MREP) x (16*NREP) tile: MREP*NREP v_wmma per k-step,
// reusing each A fragment NREP times and each B fragment MREP times (halves L2
// traffic at <2,2> vs the naive 16x16-per-wave version).
// Optional per-row mask: row m zeroed if t >= declen[m].
// K multiple of 32; M multiple of 16*MREP; N multiple of 16*NREP.
// A fragment per ISA layout: lanes 0-15 -> row=lane,    K = {0..7, 16..23}+k0
//                            lanes16-31 -> row=lane-16, K = {8..15, 24..31}+k0
// B fragment: lane holds column n0+(lane&15); K same as A fragment ({0..7,16..23} / {8..15,24..31}).
// ---------------------------------------------------------------------------
template<int MREP, int NREP>
__global__ void gemm_wmma_rb(const _Float16* __restrict__ A, int lda,
                             const _Float16* __restrict__ Bt, int ldb,
                             float* __restrict__ C, int ldc,
                             const float* __restrict__ bias,
                             int M, int N, int K,
                             const int* __restrict__ declen, int t)
{
    const int wave = threadIdx.x >> 5;
    const int lane = threadIdx.x & 31;
    constexpr int MT = 16 * MREP;
    constexpr int NT = 16 * NREP;
    const int ntiles = N / NT;
    const int tile = blockIdx.x * (blockDim.x >> 5) + wave;
    if (tile >= (M / MT) * ntiles) return;

    const int m0 = (tile / ntiles) * MT;
    const int n0 = (tile % ntiles) * NT;
    const int l15 = lane & 15;
    const bool hi = (lane >= 16);

    const _Float16* Ap[MREP];
    const _Float16* Bp[NREP];
#pragma unroll
    for (int i = 0; i < MREP; ++i)
        Ap[i] = A + (size_t)(m0 + 16 * i + l15) * lda + (hi ? 8 : 0);
#pragma unroll
    for (int j = 0; j < NREP; ++j)
        Bp[j] = Bt + (size_t)(n0 + 16 * j + l15) * ldb + (hi ? 8 : 0);

    v8f acc[MREP][NREP] = {};

    for (int k0 = 0; k0 < K; k0 += 32) {
        v16h av[MREP], bv[NREP];
#pragma unroll
        for (int i = 0; i < MREP; ++i) {
            h8 lo = *(const h8*)(Ap[i] + k0);
            h8 hh = *(const h8*)(Ap[i] + k0 + 16);
#pragma unroll
            for (int q = 0; q < 8; ++q) { av[i][q] = lo[q]; av[i][8 + q] = hh[q]; }
        }
#pragma unroll
        for (int j = 0; j < NREP; ++j) {
            h8 lo = *(const h8*)(Bp[j] + k0);
            h8 hh = *(const h8*)(Bp[j] + k0 + 16);
#pragma unroll
            for (int q = 0; q < 8; ++q) { bv[j][q] = lo[q]; bv[j][8 + q] = hh[q]; }
        }
#pragma unroll
        for (int i = 0; i < MREP; ++i)
#pragma unroll
            for (int j = 0; j < NREP; ++j)
                acc[i][j] = __builtin_amdgcn_wmma_f32_16x16x32_f16(
                    false, av[i], false, bv[j], (short)0, acc[i][j], false, false);
    }

#pragma unroll
    for (int j = 0; j < NREP; ++j) {
        const int n = n0 + 16 * j + l15;
        const float bb = bias ? bias[n] : 0.0f;
#pragma unroll
        for (int i = 0; i < MREP; ++i) {
#pragma unroll
            for (int v = 0; v < 8; ++v) {
                int m = m0 + 16 * i + v + (hi ? 8 : 0);
                float val = acc[i][j][v] + bb;
                if (declen && t >= declen[m]) val = 0.0f;
                C[(size_t)m * ldc + n] = val;
            }
        }
    }
}

// ---------------------------------------------------------------------------
// Setup kernels
// ---------------------------------------------------------------------------

// Stable descending argsort of lengths (B=64), one block of 64 threads.
// Also emits sorted_caps / decode_lengths / sort_idx (as floats) to d_out.
__global__ void prep_kernel(const int* __restrict__ captions,
                            const int* __restrict__ cap_lens,
                            int* __restrict__ sidx_ws, int* __restrict__ declen_ws,
                            float* __restrict__ out_caps,
                            float* __restrict__ out_declen,
                            float* __restrict__ out_sidx)
{
    __shared__ int len[BATCH];
    __shared__ int sidx_sh[BATCH];
    int i = threadIdx.x;
    len[i] = cap_lens[i];
    __syncthreads();
    int li = len[i], rank = 0;
    for (int j = 0; j < BATCH; ++j) {
        int lj = len[j];
        if (lj > li || (lj == li && j < i)) rank++;
    }
    sidx_sh[rank] = i;
    __syncthreads();
    int s = sidx_sh[i];
    sidx_ws[i] = s;
    out_sidx[i] = (float)s;
    int dl = len[s] - 1;
    declen_ws[i] = dl;
    out_declen[i] = (float)dl;
    for (int j = 0; j < LCAP; ++j)
        out_caps[i * LCAP + j] = (float)captions[s * LCAP + j];
}

// enc_f16[b][p][k] = (f16) encoder_out[sidx[b]][p][k]
__global__ void gather_enc_kernel(const float* __restrict__ enc,
                                  const int* __restrict__ sidx,
                                  _Float16* __restrict__ dst)
{
    size_t idx = (size_t)blockIdx.x * blockDim.x + threadIdx.x;
    int b = (int)(idx / ((size_t)SPOS * ENCD));
    size_t r = idx % ((size_t)SPOS * ENCD);
    dst[idx] = (_Float16)enc[(size_t)sidx[b] * SPOS * ENCD + r];
}

// mean_f16[b][k] = mean_p encoder_out[sidx[b]][p][k]  (f32 accumulate)
__global__ void mean_kernel(const float* __restrict__ enc,
                            const int* __restrict__ sidx,
                            _Float16* __restrict__ mean_h)
{
    int idx = blockIdx.x * blockDim.x + threadIdx.x;   // 64*2048
    int b = idx >> 11, k = idx & 2047;
    const float* base = enc + (size_t)sidx[b] * SPOS * ENCD + k;
    float s = 0.0f;
    for (int p = 0; p < SPOS; ++p) s += base[(size_t)p * ENCD];
    mean_h[idx] = (_Float16)(s * (1.0f / (float)SPOS));
}

// emb_f16[b][j][e] = (f16) embedding[captions[sidx[b]][j]][e]
__global__ void emb_kernel(const int* __restrict__ captions,
                           const int* __restrict__ sidx,
                           const float* __restrict__ embedding,
                           _Float16* __restrict__ emb_h)
{
    int idx = blockIdx.x * blockDim.x + threadIdx.x;   // 64*20*512
    int b = idx / (LCAP * EDIM);
    int r = idx % (LCAP * EDIM);
    int j = r >> 9, e = r & 511;
    int cap = captions[sidx[b] * LCAP + j];
    emb_h[idx] = (_Float16)embedding[(size_t)cap * EDIM + e];
}

// Bt[n][k] = (f16) W[k][n]   (W is KxN row-major)
__global__ void pack_t_kernel(const float* __restrict__ W,
                              _Float16* __restrict__ Bt, int K, int N)
{
    size_t idx = (size_t)blockIdx.x * blockDim.x + threadIdx.x;
    if (idx >= (size_t)K * N) return;
    int n = (int)(idx / K), k = (int)(idx % K);
    Bt[idx] = (_Float16)W[(size_t)k * N + n];
}

// Fused gates weight: Btg[n][k] = k<2560 ? W_ih[n][k] : W_hh[n][k-2560]
// (both already stored as [N=2048][K] row-major -> output-n-major directly)
// also bg[n] = b_ih[n] + b_hh[n]
__global__ void pack_gates_kernel(const float* __restrict__ Wih,
                                  const float* __restrict__ Whh,
                                  const float* __restrict__ bih,
                                  const float* __restrict__ bhh,
                                  _Float16* __restrict__ Bt,
                                  float* __restrict__ bg)
{
    size_t idx = (size_t)blockIdx.x * blockDim.x + threadIdx.x;  // 2048*3072
    int n = (int)(idx / XDIM), k = (int)(idx % XDIM);
    _Float16 v = (k < 2560) ? (_Float16)Wih[(size_t)n * 2560 + k]
                            : (_Float16)Whh[(size_t)n * 512 + (k - 2560)];
    Bt[idx] = v;
    if (idx < 2048) bg[idx] = bih[idx] + bhh[idx];
}

// plain f32 -> f16 convert (W_out is [V][D] = already n-major)
__global__ void conv_f16_kernel(const float* __restrict__ src,
                                _Float16* __restrict__ dst, int n)
{
    int idx = blockIdx.x * blockDim.x + threadIdx.x;
    if (idx < n) dst[idx] = (_Float16)src[idx];
}

// x tail <- h0 (f16)
__global__ void initx_kernel(const float* __restrict__ h, _Float16* __restrict__ x)
{
    int idx = blockIdx.x * blockDim.x + threadIdx.x;   // 64*512
    int b = idx >> 9, n = idx & 511;
    x[(size_t)b * XDIM + 2560 + n] = (_Float16)h[idx];
}

// ---------------------------------------------------------------------------
// Per-step kernels
// ---------------------------------------------------------------------------

// scores[b][p] = sum_a tanh(enc_proj[b][p][a] + hproj[b][a]) * W_full[a] + b_full
__global__ void att_scores_kernel(const float* __restrict__ encproj,
                                  const float* __restrict__ hproj,
                                  const float* __restrict__ Wfull,
                                  const float* __restrict__ bfull,
                                  float* __restrict__ scores)
{
    __shared__ float red[128];
    int bp = blockIdx.x;           // 0 .. B*SPOS-1
    int b = bp / SPOS;
    int tid = threadIdx.x;
    const float* ep = encproj + (size_t)bp * ADIM;
    const float* hp = hproj + (size_t)b * ADIM;
    float s = 0.0f;
    for (int a = tid; a < ADIM; a += 128)
        s += tanhf(ep[a] + hp[a]) * Wfull[a];
    red[tid] = s; __syncthreads();
    for (int off = 64; off > 0; off >>= 1) {
        if (tid < off) red[tid] += red[tid + off];
        __syncthreads();
    }
    if (tid == 0) scores[bp] = red[0] + bfull[0];
}

// alpha = softmax over 196; also writes masked alphas into d_out[b][t][p]
__global__ void softmax_kernel(const float* __restrict__ scores,
                               float* __restrict__ alpha,
                               float* __restrict__ out_alpha,
                               const int* __restrict__ declen, int t)
{
    __shared__ float red[256];
    int b = blockIdx.x, tid = threadIdx.x;
    const float* s = scores + b * SPOS;
    float v = (tid < SPOS) ? s[tid] : -3.4e38f;
    red[tid] = v; __syncthreads();
    for (int off = 128; off > 0; off >>= 1) {
        if (tid < off) red[tid] = fmaxf(red[tid], red[tid + off]);
        __syncthreads();
    }
    float mx = red[0]; __syncthreads();
    float e = (tid < SPOS) ? __expf(v - mx) : 0.0f;
    red[tid] = e; __syncthreads();
    for (int off = 128; off > 0; off >>= 1) {
        if (tid < off) red[tid] += red[tid + off];
        __syncthreads();
    }
    float inv = 1.0f / red[0];
    if (tid < SPOS) {
        float a = e * inv;
        alpha[b * SPOS + tid] = a;
        bool m = t < declen[b];
        out_alpha[((size_t)b * TSTEP + t) * SPOS + tid] = m ? a : 0.0f;
    }
}

// ctx[b][e] = sum_p alpha[b][p] * enc_f16[b][p][e]
__global__ void ctx_kernel(const float* __restrict__ alpha,
                           const _Float16* __restrict__ enc_h,
                           float* __restrict__ ctx)
{
    __shared__ float al_s[SPOS];
    int b = blockIdx.x;
    int tid = threadIdx.x;
    if (tid < SPOS) al_s[tid] = alpha[b * SPOS + tid];
    __syncthreads();
    int e = blockIdx.y * 128 + tid;             // 2048 = 16*128
    const _Float16* base = enc_h + (size_t)b * SPOS * ENCD + e;
    float s = 0.0f;
    for (int p = 0; p < SPOS; ++p) s += al_s[p] * (float)base[(size_t)p * ENCD];
    ctx[b * ENCD + e] = s;
}

// x[b][0:512] = emb_t ; x[b][512:2560] = sigmoid(gatepre)*ctx  (f16)
__global__ void build_x_kernel(const _Float16* __restrict__ emb_h,
                               const float* __restrict__ gatepre,
                               const float* __restrict__ ctx,
                               _Float16* __restrict__ x, int t)
{
    int idx = blockIdx.x * blockDim.x + threadIdx.x;   // 64*2560
    int b = idx / 2560, j = idx % 2560;
    if (j < 512) {
        x[(size_t)b * XDIM + j] = emb_h[((size_t)b * LCAP + t) * EDIM + j];
    } else {
        int e = j - 512;
        float g = sigf(gatepre[b * ENCD + e]);
        x[(size_t)b * XDIM + j] = (_Float16)(g * ctx[b * ENCD + e]);
    }
}

// LSTM cell + masked state update; writes hnew (f16, for pred GEMM) and
// masked h into x tail (f16, for next-step GEMMs) + f32 h/c states.
__global__ void lstm_kernel(const float* __restrict__ gates,
                            float* __restrict__ h, float* __restrict__ c,
                            _Float16* __restrict__ hnew_h,
                            _Float16* __restrict__ x,
                            const int* __restrict__ declen, int t)
{
    int idx = blockIdx.x * blockDim.x + threadIdx.x;   // 64*512
    int b = idx >> 9, n = idx & 511;
    const float* g = gates + (size_t)b * 2048;
    float i_ = sigf(g[n]);
    float f_ = sigf(g[512 + n]);
    float gg = tanhf(g[1024 + n]);
    float o_ = sigf(g[1536 + n]);
    float cn = f_ * c[idx] + i_ * gg;
    float hn = o_ * tanhf(cn);
    bool m = t < declen[b];
    float h2 = m ? hn : h[idx];
    float c2 = m ? cn : c[idx];
    h[idx] = h2; c[idx] = c2;
    hnew_h[idx] = (_Float16)hn;
    x[(size_t)b * XDIM + 2560 + n] = (_Float16)h2;
}

// ---------------------------------------------------------------------------
// Host side
// ---------------------------------------------------------------------------
static inline void launch_gemm(const _Float16* A, int lda, const _Float16* Bt, int ldb,
                               float* C, int ldc, const float* bias,
                               int M, int N, int K, const int* declen, int t,
                               hipStream_t s)
{
    // M is always a multiple of 32 here (64 or 12544).
    if ((N & 31) == 0) {
        int tiles = (M / 32) * (N / 32);
        gemm_wmma_rb<2, 2><<<(tiles + 7) / 8, 256, 0, s>>>(A, lda, Bt, ldb, C, ldc,
                                                           bias, M, N, K, declen, t);
    } else {
        int tiles = (M / 32) * (N / 16);
        gemm_wmma_rb<2, 1><<<(tiles + 7) / 8, 256, 0, s>>>(A, lda, Bt, ldb, C, ldc,
                                                           bias, M, N, K, declen, t);
    }
}

extern "C" void kernel_launch(void* const* d_in, const int* in_sizes, int n_in,
                              void* d_out, int out_size, void* d_ws, size_t ws_size,
                              hipStream_t stream)
{
    // ---- inputs (setup_inputs dict order) ----
    const float* encoder_out = (const float*)d_in[0];
    const int*   captions    = (const int*)d_in[1];
    const int*   cap_lens    = (const int*)d_in[2];
    const float* embedding   = (const float*)d_in[3];
    const float* W_enc_att   = (const float*)d_in[4];
    const float* b_enc_att   = (const float*)d_in[5];
    const float* W_dec_att   = (const float*)d_in[6];
    const float* b_dec_att   = (const float*)d_in[7];
    const float* W_full_att  = (const float*)d_in[8];
    const float* b_full_att  = (const float*)d_in[9];
    const float* W_init_h    = (const float*)d_in[10];
    const float* b_init_h    = (const float*)d_in[11];
    const float* W_init_c    = (const float*)d_in[12];
    const float* b_init_c    = (const float*)d_in[13];
    const float* W_beta      = (const float*)d_in[14];
    const float* b_beta      = (const float*)d_in[15];
    const float* W_ih        = (const float*)d_in[16];
    const float* b_ih        = (const float*)d_in[17];
    const float* W_hh        = (const float*)d_in[18];
    const float* b_hh        = (const float*)d_in[19];
    const float* W_out       = (const float*)d_in[20];
    const float* b_out       = (const float*)d_in[21];

    // ---- outputs (tuple concatenated, f32) ----
    float* out = (float*)d_out;
    const size_t OFF_PREDS = 0;                                   // 64*19*10000
    const size_t OFF_CAPS  = OFF_PREDS + (size_t)BATCH * TSTEP * VOC;
    const size_t OFF_DECL  = OFF_CAPS + (size_t)BATCH * LCAP;
    const size_t OFF_ALPH  = OFF_DECL + BATCH;
    const size_t OFF_SIDX  = OFF_ALPH + (size_t)BATCH * TSTEP * SPOS;
    float* out_preds  = out + OFF_PREDS;
    float* out_caps   = out + OFF_CAPS;
    float* out_declen = out + OFF_DECL;
    float* out_alpha  = out + OFF_ALPH;
    float* out_sidx   = out + OFF_SIDX;

    // ---- workspace carving (256B aligned) ----
    char* ws = (char*)d_ws;
    size_t off = 0;
    auto carve = [&](size_t bytes) -> void* {
        void* p = ws + off;
        off += (bytes + 255) & ~(size_t)255;
        return p;
    };
    int*       sidx     = (int*)carve(BATCH * sizeof(int));
    int*       declen   = (int*)carve(BATCH * sizeof(int));
    _Float16*  enc_h    = (_Float16*)carve((size_t)BATCH * SPOS * ENCD * 2);   // 51.4 MB
    float*     encproj  = (float*)carve((size_t)BATCH * SPOS * ADIM * 4);      // 25.7 MB
    _Float16*  Wenc_t   = (_Float16*)carve((size_t)ADIM * ENCD * 2);
    _Float16*  Winh_t   = (_Float16*)carve((size_t)DDIM * ENCD * 2);
    _Float16*  Winc_t   = (_Float16*)carve((size_t)DDIM * ENCD * 2);
    _Float16*  Wdec_t   = (_Float16*)carve((size_t)ADIM * DDIM * 2);
    _Float16*  Wbeta_t  = (_Float16*)carve((size_t)ENCD * DDIM * 2);
    _Float16*  Wgates_t = (_Float16*)carve((size_t)2048 * XDIM * 2);           // 12.6 MB
    float*     bgates   = (float*)carve(2048 * 4);
    _Float16*  Wout_h   = (_Float16*)carve((size_t)VOC * DDIM * 2);            // 10.2 MB
    _Float16*  mean_h   = (_Float16*)carve((size_t)BATCH * ENCD * 2);
    _Float16*  emb_h    = (_Float16*)carve((size_t)BATCH * LCAP * EDIM * 2);
    float*     h_f32    = (float*)carve((size_t)BATCH * DDIM * 4);
    float*     c_f32    = (float*)carve((size_t)BATCH * DDIM * 4);
    _Float16*  x_h      = (_Float16*)carve((size_t)BATCH * XDIM * 2);
    _Float16*  hnew_h   = (_Float16*)carve((size_t)BATCH * DDIM * 2);
    float*     hproj    = (float*)carve((size_t)BATCH * ADIM * 4);
    float*     gatepre  = (float*)carve((size_t)BATCH * ENCD * 4);
    float*     gates    = (float*)carve((size_t)BATCH * 2048 * 4);
    float*     scores   = (float*)carve((size_t)BATCH * SPOS * 4);
    float*     alpha    = (float*)carve((size_t)BATCH * SPOS * 4);
    float*     ctx      = (float*)carve((size_t)BATCH * ENCD * 4);
    (void)ws_size; (void)in_sizes; (void)n_in; (void)out_size;

    // ---- one-time setup ----
    prep_kernel<<<1, BATCH, 0, stream>>>(captions, cap_lens, sidx, declen,
                                         out_caps, out_declen, out_sidx);

    {
        size_t n = (size_t)BATCH * SPOS * ENCD;
        gather_enc_kernel<<<(unsigned)((n + 255) / 256), 256, 0, stream>>>(encoder_out, sidx, enc_h);
    }
    mean_kernel<<<(BATCH * ENCD) / 256, 256, 0, stream>>>(encoder_out, sidx, mean_h);
    emb_kernel<<<(BATCH * LCAP * EDIM) / 256, 256, 0, stream>>>(captions, sidx, embedding, emb_h);

    pack_t_kernel<<<(ENCD * ADIM) / 256, 256, 0, stream>>>(W_enc_att, Wenc_t, ENCD, ADIM);
    pack_t_kernel<<<(ENCD * DDIM) / 256, 256, 0, stream>>>(W_init_h, Winh_t, ENCD, DDIM);
    pack_t_kernel<<<(ENCD * DDIM) / 256, 256, 0, stream>>>(W_init_c, Winc_t, ENCD, DDIM);
    pack_t_kernel<<<(DDIM * ADIM) / 256, 256, 0, stream>>>(W_dec_att, Wdec_t, DDIM, ADIM);
    pack_t_kernel<<<(DDIM * ENCD) / 256, 256, 0, stream>>>(W_beta, Wbeta_t, DDIM, ENCD);
    pack_gates_kernel<<<(2048 * XDIM) / 256, 256, 0, stream>>>(W_ih, W_hh, b_ih, b_hh, Wgates_t, bgates);
    conv_f16_kernel<<<(VOC * DDIM) / 256, 256, 0, stream>>>(W_out, Wout_h, VOC * DDIM);

    // enc_proj = enc_f16 @ W_enc_att + b   (12544 x 512, K=2048)  -- big WMMA GEMM
    launch_gemm(enc_h, ENCD, Wenc_t, ENCD, encproj, ADIM, b_enc_att,
                BATCH * SPOS, ADIM, ENCD, nullptr, 0, stream);
    // h0, c0 = mean @ W_init_{h,c} + b   (64 x 512, K=2048)
    launch_gemm(mean_h, ENCD, Winh_t, ENCD, h_f32, DDIM, b_init_h,
                BATCH, DDIM, ENCD, nullptr, 0, stream);
    launch_gemm(mean_h, ENCD, Winc_t, ENCD, c_f32, DDIM, b_init_c,
                BATCH, DDIM, ENCD, nullptr, 0, stream);
    initx_kernel<<<(BATCH * DDIM) / 256, 256, 0, stream>>>(h_f32, x_h);

    const _Float16* h_in_x = x_h + 2560;   // A-view of h inside x (lda = XDIM)

    // ---- sequential decode ----
    for (int t = 0; t < TSTEP; ++t) {
        // hproj = h @ W_dec_att + b   (64 x 512, K=512)
        launch_gemm(h_in_x, XDIM, Wdec_t, DDIM, hproj, ADIM, b_dec_att,
                    BATCH, ADIM, DDIM, nullptr, 0, stream);
        // attention scores + softmax
        att_scores_kernel<<<BATCH * SPOS, 128, 0, stream>>>(encproj, hproj,
                                                            W_full_att, b_full_att, scores);
        softmax_kernel<<<BATCH, 256, 0, stream>>>(scores, alpha, out_alpha, declen, t);
        // ctx = alpha @ enc
        ctx_kernel<<<dim3(BATCH, ENCD / 128), 128, 0, stream>>>(alpha, enc_h, ctx);
        // gate_pre = h @ W_beta + b   (64 x 2048, K=512)
        launch_gemm(h_in_x, XDIM, Wbeta_t, DDIM, gatepre, ENCD, b_beta,
                    BATCH, ENCD, DDIM, nullptr, 0, stream);
        // x = [emb_t | sigmoid(gate_pre)*ctx | h]
        build_x_kernel<<<(BATCH * 2560) / 256, 256, 0, stream>>>(emb_h, gatepre, ctx, x_h, t);
        // gates = x @ [W_ih;W_hh]^T + (b_ih+b_hh)   (64 x 2048, K=3072)
        launch_gemm(x_h, XDIM, Wgates_t, XDIM, gates, 2048, bgates,
                    BATCH, 2048, XDIM, nullptr, 0, stream);
        // LSTM cell
        lstm_kernel<<<(BATCH * DDIM) / 256, 256, 0, stream>>>(gates, h_f32, c_f32,
                                                              hnew_h, x_h, declen, t);
        // preds[:, t, :] = mask * (h_new @ W_out^T + b_out)   (64 x 10000, K=512)
        launch_gemm(hnew_h, DDIM, Wout_h, DDIM, out_preds + (size_t)t * VOC,
                    TSTEP * VOC, b_out, BATCH, VOC, DDIM, declen, t, stream);
    }
}
